// ACoef_36558761624427
// MI455X (gfx1250) — matrix operand order repaired and checked by reading, not compiled, not run on hardware
//
#include <hip/hip_runtime.h>

typedef __bf16 v16bf __attribute__((ext_vector_type(16)));
typedef float  v8f   __attribute__((ext_vector_type(8)));

namespace {
constexpr int kN   = 224;   // matrix dim
constexpr int kPad = 225;   // LDS row stride (odd -> conflict-free column walks)
constexpr int kNT  = 14;    // 224/16 tiles per dimension
constexpr int kBlk = 256;   // 8 wave32 per workgroup
}

__device__ __forceinline__ v8f wmma_bf16(v16bf a, v16bf b, v8f c) {
  // D = A*B + C, f32 accumulate (v_wmma_f32_16x16x32_bf16)
  return __builtin_amdgcn_wmma_f32_16x16x32_bf16(
      /*neg_a=*/false, a, /*neg_b=*/false, b,
      /*c_mod=*/(short)0, c, /*reuse_a=*/false, /*reuse_b=*/false);
}

// Split 16 f32 lane-values into bf16 hi + bf16 lo fragments (packed 2/dword).
__device__ __forceinline__ void pack_split(const float v[16], v16bf& hi, v16bf& lo) {
  union Cvt { unsigned u[8]; v16bf b; };
  Cvt H, L;
#pragma unroll
  for (int d = 0; d < 8; ++d) {
    const float a = v[2 * d];
    const float b = v[2 * d + 1];
    const unsigned ha = __float_as_uint(a) & 0xffff0000u;  // bf16(a) by truncation
    const unsigned hb = __float_as_uint(b) & 0xffff0000u;
    const float ra = a - __uint_as_float(ha);              // exact residual
    const float rb = b - __uint_as_float(hb);
    H.u[d] = (ha >> 16) | hb;
    L.u[d] = (__float_as_uint(ra) >> 16) | (__float_as_uint(rb) & 0xffff0000u);
  }
  hi = H.b;
  lo = L.b;
}

__global__ __launch_bounds__(kBlk)
void acoef_kernel(const float* __restrict__ x, const float* __restrict__ coef,
                  float* __restrict__ out) {
  extern __shared__ float smem[];
  float* P2  = smem;                 // [224][225] f32: x^2, LDS-resident
  float* dg  = smem + kN * kPad;     // [4][224]   diag(x^2..x^5)
  float* red = dg + 4 * kN;          // [256]      final reduction scratch

  const int b    = blockIdx.x;
  const int tid  = threadIdx.x;
  const int wave = tid >> 5;
  const int lane = tid & 31;
  const int lm   = lane & 15;        // M/N index within tile
  const int hw   = lane >> 4;        // half-wave

  const float* __restrict__ X = x + (size_t)b * (kN * kN);

  // Pull this batch's 200KB of x toward L2 (global_prefetch_b8).
  for (int off = tid * 64; off < kN * kN * 4; off += kBlk * 64)
    __builtin_prefetch((const char*)X + off, 0, 1);

  // ---------------- Phase 1: P2 = X @ X  (split-bf16 WMMA) -> LDS ----------------
  for (int t = wave; t < kNT * kNT; t += kBlk / 32) {
    const int ti = t / kNT, tj = t % kNT;
    const int m0 = ti * 16, n0 = tj * 16;
    const float* Arow = X + (size_t)(m0 + lm) * kN;
    const int ncol = n0 + lm;
    v8f acc = {0.f, 0.f, 0.f, 0.f, 0.f, 0.f, 0.f, 0.f};
    for (int k0 = 0; k0 < kN; k0 += 32) {
      // A fragment: rows of X; element e -> K = k0 + e + 8*hw + (e>=8 ? 8 : 0)
      float av[16];
      {
        const float4* q0 = reinterpret_cast<const float4*>(Arow + k0 + 8 * hw);
        const float4* q1 = reinterpret_cast<const float4*>(Arow + k0 + 16 + 8 * hw);
        float4 t0 = q0[0], t1 = q0[1], t2 = q1[0], t3 = q1[1];
        av[0]=t0.x; av[1]=t0.y; av[2]=t0.z;  av[3]=t0.w;
        av[4]=t1.x; av[5]=t1.y; av[6]=t1.z;  av[7]=t1.w;
        av[8]=t2.x; av[9]=t2.y; av[10]=t2.z; av[11]=t2.w;
        av[12]=t3.x; av[13]=t3.y; av[14]=t3.z; av[15]=t3.w;
      }
      // B fragment: columns of X; element e -> K = k0 + e + 8*hw + (e>=8 ? 8 : 0), N = ncol
      float bv[16];
#pragma unroll
      for (int e = 0; e < 16; ++e)
        bv[e] = X[(size_t)(k0 + e + 8 * hw + (e >= 8 ? 8 : 0)) * kN + ncol];
      v16bf ahi, alo, bhi, blo;
      pack_split(av, ahi, alo);
      pack_split(bv, bhi, blo);
      acc = wmma_bf16(ahi, bhi, acc);
      acc = wmma_bf16(ahi, blo, acc);
      acc = wmma_bf16(alo, bhi, acc);
    }
    // C/D layout: element r -> row m0 + r + 8*hw, col ncol
#pragma unroll
    for (int r = 0; r < 8; ++r)
      P2[(size_t)(m0 + r + 8 * hw) * kPad + ncol] = acc[r];
  }
  __syncthreads();

  // ---------------- Phase 2: diag(x^2), diag(x^3), diag(x^4) (f32 VALU) ----------------
  if (tid < kN) {
    const int n = tid;
    const float* rowp = P2 + (size_t)n * kPad;
    float d3 = 0.f, d4 = 0.f;
    for (int k = 0; k < kN; ++k) {
      const float p = rowp[k];
      d3 += p * X[(size_t)k * kN + n];      // <P2[n,:], x[:,n]>
      d4 += p * P2[(size_t)k * kPad + n];   // <P2[n,:], P2[:,n]>
    }
    dg[0 * kN + n] = rowp[n];
    dg[1 * kN + n] = d3;
    dg[2 * kN + n] = d4;
  }

  // ---------------- Phase 3: diag(x^5): P3 = P2 @ X tiles, consumed on the fly ----------
  for (int ti = wave; ti < kNT; ti += kBlk / 32) {
    const int m0 = ti * 16;
    const float* Arow = P2 + (size_t)(m0 + lm) * kPad;
    float dpart[8];
#pragma unroll
    for (int r = 0; r < 8; ++r) dpart[r] = 0.f;
    for (int tj = 0; tj < kNT; ++tj) {
      const int n0 = tj * 16;
      const int ncol = n0 + lm;
      v8f acc = {0.f, 0.f, 0.f, 0.f, 0.f, 0.f, 0.f, 0.f};
      for (int k0 = 0; k0 < kN; k0 += 32) {
        float av[16];
#pragma unroll
        for (int e = 0; e < 16; ++e)
          av[e] = Arow[k0 + 8 * hw + e + (e >= 8 ? 8 : 0)];   // ds_load from P2
        float bv[16];
#pragma unroll
        for (int e = 0; e < 16; ++e)
          bv[e] = X[(size_t)(k0 + e + 8 * hw + (e >= 8 ? 8 : 0)) * kN + ncol];
        v16bf ahi, alo, bhi, blo;
        pack_split(av, ahi, alo);
        pack_split(bv, bhi, blo);
        acc = wmma_bf16(ahi, bhi, acc);
        acc = wmma_bf16(ahi, blo, acc);
        acc = wmma_bf16(alo, bhi, acc);
      }
      // diag5[m] += sum_col P3[m][col] * P2[col][m]; P2 pad stride keeps this conflict-free
#pragma unroll
      for (int r = 0; r < 8; ++r)
        dpart[r] += acc[r] * P2[(size_t)ncol * kPad + (m0 + r + 8 * hw)];
    }
#pragma unroll
    for (int r = 0; r < 8; ++r) {
      float s = dpart[r];
      s += __shfl_xor(s, 1);
      s += __shfl_xor(s, 2);
      s += __shfl_xor(s, 4);
      s += __shfl_xor(s, 8);   // sum over the 16 lanes of each half-wave
      if (lm == 0) dg[3 * kN + m0 + r + 8 * hw] = s;
    }
  }
  __syncthreads();

  // ---------------- Phase 4: traces * coef / (N^2)^(i+j+2), block-reduce --------------
  float part = 0.f;
  if (tid < kN) {
    double inv[9];
    inv[0] = 1.0;
#pragma unroll
    for (int k = 1; k < 9; ++k) inv[k] = inv[k - 1] * (1.0 / (double)(kN * kN));
#pragma unroll
    for (int i = 0; i < 4; ++i) {
      const float d = dg[i * kN + tid];
      float dp = d;
#pragma unroll
      for (int j = 0; j < 4; ++j) {
        part += coef[i * 4 + j] * dp * (float)inv[i + j + 2];
        dp *= d;
      }
    }
  }
  red[tid] = part;
  __syncthreads();
  if (tid < 32) {
    float s = 0.f;
#pragma unroll
    for (int j = 0; j < 8; ++j) s += red[tid + 32 * j];
    s += __shfl_xor(s, 16);
    s += __shfl_xor(s, 8);
    s += __shfl_xor(s, 4);
    s += __shfl_xor(s, 2);
    s += __shfl_xor(s, 1);
    if (tid == 0) out[b] = s;
  }
}

extern "C" void kernel_launch(void* const* d_in, const int* in_sizes, int n_in,
                              void* d_out, int out_size, void* d_ws, size_t ws_size,
                              hipStream_t stream) {
  const float* x    = (const float*)d_in[0];   // [B, 224, 224] f32
  const float* coef = (const float*)d_in[1];   // [4, 4] f32
  float* out        = (float*)d_out;           // [B] f32
  const int B = in_sizes[0] / (kN * kN);
  const size_t lds_bytes = (size_t)(kN * kPad + 4 * kN + kBlk) * sizeof(float);
  acoef_kernel<<<dim3(B), dim3(kBlk), lds_bytes, stream>>>(x, coef, out);
}
